// GNN_75153337745817
// MI455X (gfx1250) — matrix operand order repaired and checked by reading, not compiled, not run on hardware
//
#include <hip/hip_runtime.h>

typedef _Float16 f16;
typedef __attribute__((ext_vector_type(16))) _Float16 v16h;
typedef __attribute__((ext_vector_type(8)))  _Float16 v8h;
typedef __attribute__((ext_vector_type(8)))  float    v8f;

#define NNODES   50000
#define SEQL     16
#define EMBD     64
#define HIDD     64
#define STATE    70
#define NEDGES   800000
#define NTGT     8192
#define KP       96      // padded K for x / out rows
#define CATK     160     // padded K for [x, agg] rows
#define OPAD     80      // padded output columns (5 tiles of 16)

// ---------------- workspace layout (bytes, all 256-aligned) ----------------
#define OFF_WIH_MAP  0u
#define OFF_WHH_MAP  24576u
#define OFF_WIH_MEM  49152u
#define OFF_WHH_MEM  73728u
#define OFF_WMSGT    98304u      // 3*80*96 f16 = 46080
#define OFF_WUPDT    144384u     // 80*160 f16 = 25600
#define OFF_WLIN1T   169984u     // 80*96 f16 = 15360
#define OFF_X96      185344u     // N*96  f16 = 9600000
#define OFF_CAT      9785344u    // N*160 f16 = 16000000
#define OFF_AGG      25785344u   // N*80  u32 = 16000000
#define OFF_OUT96    41785344u   // N*96  f16 = 9600000

static __device__ inline v8f vzero8() {
  v8f z = {0.f,0.f,0.f,0.f,0.f,0.f,0.f,0.f};
  return z;
}

static __device__ inline v8f wmma32(v16h a, v16h b, v8f c) {
  return __builtin_amdgcn_wmma_f32_16x16x32_f16(false, a, false, b, (short)0, c,
                                                false, false);
}

// ---- fast transcendentals: map straight onto V_EXP_F32 / V_RCP_F32 / V_TANH ----
#define LOG2E_F 1.44269504088896340736f

static __device__ inline float fast_exp2(float x) {
#if __has_builtin(__builtin_amdgcn_exp2f)
  return __builtin_amdgcn_exp2f(x);
#else
  return exp2f(x);
#endif
}
static __device__ inline float fast_rcp(float x) {
#if __has_builtin(__builtin_amdgcn_rcpf)
  return __builtin_amdgcn_rcpf(x);
#else
  return 1.f / x;
#endif
}
static __device__ inline float fast_sigmoid(float x) {
  // 1 / (1 + exp(-x)) == rcp(1 + exp2(-x*log2e)) : one v_exp + one v_rcp
  return fast_rcp(1.f + fast_exp2(-x * LOG2E_F));
}
static __device__ inline float fast_tanh(float x) {
#if __has_builtin(__builtin_amdgcn_tanhf)
  return __builtin_amdgcn_tanhf(x);
#else
  // tanh(x) = 1 - 2/(exp(2x)+1) : one v_exp + one v_rcp (monotone, saturates)
  float e = fast_exp2(x * (2.f * LOG2E_F));
  return 1.f - 2.f * fast_rcp(e + 1.f);
#endif
}

// A fragment (16x32 f16, MxK) from LDS; `stride` in halves (multiple of 8).
static __device__ inline v16h load_a_lds(const f16* base, int stride, int kc, int lane) {
  const int m  = lane & 15;
  const int k0 = kc * 32 + ((lane < 16) ? 0 : 16);
  const f16* p = base + m * stride + k0;
  union { v16h v; v8h h[2]; } u;
  u.h[0] = *(const v8h*)(p);        // K = k0 .. k0+7
  u.h[1] = *(const v8h*)(p + 8);    // K = k0+8 .. k0+15
  return u.v;
}

// B fragment (32x16 f16, KxN) from global W[o][k] row-major; stride halves (mult of 16).
static __device__ inline v16h load_b_glb(const f16* W, int stride, int colTile, int kc, int lane) {
  const int n = lane & 15;
  const f16* p = W + (colTile * 16 + n) * stride + kc * 32 + ((lane < 16) ? 0 : 16);
  return *(const v16h*)(p);
}

static __device__ inline unsigned ordEncode(float f) {
  unsigned u = __float_as_uint(f);
  return (u & 0x80000000u) ? ~u : (u | 0x80000000u);
}
static __device__ inline float ordDecode(unsigned e) {
  unsigned u = (e & 0x80000000u) ? (e & 0x7FFFFFFFu) : ~e;
  return __uint_as_float(u);
}

// ---------------- weight convert / transpose / pad to f16 ----------------
__global__ __launch_bounds__(256) void prep_kernel(
    const float* Wih_map, const float* Whh_map,
    const float* Wih_mem, const float* Whh_mem,
    const float* Wmsg, const float* Wupd, const float* Wlin1,
    f16* gWih_map, f16* gWhh_map, f16* gWih_mem, f16* gWhh_mem,
    f16* gWmsgT, f16* gWupdT, f16* gWlin1T)
{
  int i = blockIdx.x * 256 + threadIdx.x;
  if (i < 12288) { gWih_map[i] = (f16)Wih_map[i]; return; }
  i -= 12288;
  if (i < 12288) { gWhh_map[i] = (f16)Whh_map[i]; return; }
  i -= 12288;
  if (i < 12288) { gWih_mem[i] = (f16)Wih_mem[i]; return; }
  i -= 12288;
  if (i < 12288) { gWhh_mem[i] = (f16)Whh_mem[i]; return; }
  i -= 12288;
  if (i < 3 * OPAD * KP) {
    int e = i / (OPAD * KP), r = i % (OPAD * KP), o = r / KP, k = r % KP;
    float v = (o < STATE && k < STATE) ? Wmsg[e * STATE * STATE + k * STATE + o] : 0.f;
    gWmsgT[i] = (f16)v; return;
  }
  i -= 3 * OPAD * KP;
  if (i < OPAD * CATK) {
    int o = i / CATK, k = i % CATK;
    float v = (o < STATE && k < 2 * STATE) ? Wupd[k * STATE + o] : 0.f;
    gWupdT[i] = (f16)v; return;
  }
  i -= OPAD * CATK;
  if (i < OPAD * KP) {
    int o = i / KP, k = i % KP;
    float v = (o < STATE && k < STATE) ? Wlin1[k * STATE + o] : 0.f;
    gWlin1T[i] = (f16)v; return;
  }
}

__global__ __launch_bounds__(256) void init_agg_kernel(unsigned* agg) {
  int i = blockIdx.x * 256 + threadIdx.x;
  if (i < NNODES * OPAD) agg[i] = 0u;   // encoded "below -inf"
}

// ---------------- GRU encode: 1 block = 16 nodes, 8 waves ----------------
// waves 0..3 -> map GRU, waves 4..7 -> mem GRU; wave owns gate tiles w, w+4, w+8
__global__ __launch_bounds__(256) void encode_kernel(
    const int* tokens, const int* node_type, const float* embed,
    const f16* Wih_map, const f16* Whh_map, const f16* Wih_mem, const f16* Whh_mem,
    const float* bih_map, const float* bhh_map,
    const float* bih_mem, const float* bhh_mem,
    f16* x96, f16* cat160)
{
  __shared__ __attribute__((aligned(16))) f16 s_xt[16 * 72];
  __shared__ __attribute__((aligned(16))) f16 s_h[2][16 * 72];

  const int tid  = threadIdx.x;
  const int lane = tid & 31;
  const int wave = tid >> 5;
  const int gru  = wave >> 2;
  const int wsub = wave & 3;
  const int m0   = blockIdx.x * 16;

  const f16*   Wih = gru ? Wih_mem : Wih_map;
  const f16*   Whh = gru ? Whh_mem : Whh_map;
  const float* bih = gru ? bih_mem : bih_map;
  const float* bhh = gru ? bhh_mem : bhh_map;

  // B fragments resident in registers for all 16 steps: [gate][kchunk]
  v16h Bih[3][2], Bhh[3][2];
#pragma unroll
  for (int g = 0; g < 3; ++g)
#pragma unroll
    for (int kc = 0; kc < 2; ++kc) {
      Bih[g][kc] = load_b_glb(Wih, HIDD, wsub + 4 * g, kc, lane);
      Bhh[g][kc] = load_b_glb(Whh, HIDD, wsub + 4 * g, kc, lane);
    }

  const int col = wsub * 16 + (lane & 15);   // hidden column this lane owns
  float bi[3], bh[3];
#pragma unroll
  for (int g = 0; g < 3; ++g) {
    bi[g] = bih[g * 64 + col];
    bh[g] = bhh[g * 64 + col];
  }

  for (int i = tid; i < 2 * 16 * 72; i += 256) (&s_h[0][0])[i] = (f16)0.f;

  v8f hfrag = vzero8();   // h at (m = r + (lane>=16?8:0), col)

  for (int t = 0; t < SEQL; ++t) {
    // stage x_t = embed[tokens[:, t]] as f16
    for (int i = tid; i < 16 * 64; i += 256) {
      int m = i >> 6, k = i & 63;
      int tok = tokens[(m0 + m) * SEQL + t];
      s_xt[m * 72 + k] = (f16)embed[tok * EMBD + k];
    }
    __syncthreads();

    v8f gi[3] = {vzero8(), vzero8(), vzero8()};
    v8f gh[3] = {vzero8(), vzero8(), vzero8()};
#pragma unroll
    for (int kc = 0; kc < 2; ++kc) {
      v16h ax = load_a_lds(s_xt, 72, kc, lane);
      v16h ah = load_a_lds(&s_h[gru][0], 72, kc, lane);
#pragma unroll
      for (int g = 0; g < 3; ++g) {
        gi[g] = wmma32(ax, Bih[g][kc], gi[g]);
        gh[g] = wmma32(ah, Bhh[g][kc], gh[g]);
      }
    }
    __syncthreads();   // all reads of s_h/s_xt done before writes below

#pragma unroll
    for (int r = 0; r < 8; ++r) {
      float ir = gi[0][r] + bi[0], iz = gi[1][r] + bi[1], in = gi[2][r] + bi[2];
      float hr = gh[0][r] + bh[0], hz = gh[1][r] + bh[1], hn = gh[2][r] + bh[2];
      float rg = fast_sigmoid(ir + hr);
      float zg = fast_sigmoid(iz + hz);
      float ng = fast_tanh(in + rg * hn);
      float hnew = (1.f - zg) * ng + zg * hfrag[r];
      hfrag[r] = hnew;
      int m = r + ((lane >= 16) ? 8 : 0);
      s_h[gru][m * 72 + col] = (f16)hnew;
    }
  }
  __syncthreads();

  // build x = [one_hot(type,6), h_select] padded to 96, plus cat cols 0..69/140..159
  for (int i = tid; i < 16 * KP; i += 256) {
    int m = i / KP, k = i % KP;
    int node = m0 + m;
    int nt = node_type[node];
    float v = 0.f;
    if (k < 6) v = (nt == k) ? 1.f : 0.f;
    else if (k < STATE) {
      if (nt == 0)      v = (float)s_h[0][m * 72 + (k - 6)];
      else if (nt == 5) v = (float)s_h[1][m * 72 + (k - 6)];
    }
    f16 hv = (f16)v;
    x96[node * KP + k] = hv;
    if (k < STATE) cat160[node * CATK + k] = hv;
  }
  for (int i = tid; i < 16 * 20; i += 256) {
    int m = i / 20, k = 140 + (i % 20);
    cat160[(m0 + m) * CATK + k] = (f16)0.f;
  }
}

// ---------------- edge messages + segment-max: 1 block = 16 edges, 5 waves ----
__global__ __launch_bounds__(160) void msg_kernel(
    const f16* x96, const f16* WmsgT, const float* bmsg,
    const int* edges, unsigned* agg)
{
  __shared__ __attribute__((aligned(16))) f16 s_rows[16 * 104];
  __shared__ int s_src[16];
  __shared__ int s_dst[16];

  const int ET = NEDGES / 16;   // 50000 tiles per direction
  int bi = blockIdx.x;
  int mode, tile;
  if (bi < ET)          { mode = 0; tile = bi; }
  else if (bi < 2 * ET) { mode = 1; tile = bi - ET; }
  else                  { mode = 2; tile = bi - 2 * ET; }

  const int* esrc = edges;
  const int* edst = edges + NEDGES;
  const int tid = threadIdx.x, lane = tid & 31, wave = tid >> 5;

  if (tid < 16) {
    int e = tile * 16 + tid;
    int a, d;
    if (mode == 0)      { a = esrc[e]; d = edst[e]; }
    else if (mode == 1) { a = edst[e]; d = esrc[e]; }
    else                { a = e;       d = e;       }
    s_src[tid] = a;
    s_dst[tid] = d;
  }
  __syncthreads();
  for (int i = tid; i < 16 * KP; i += 160) {
    int m = i / KP, k = i % KP;
    s_rows[m * 104 + k] = x96[s_src[m] * KP + k];
  }
  __syncthreads();

  v8f acc = vzero8();
#pragma unroll
  for (int kc = 0; kc < 3; ++kc) {
    v16h a = load_a_lds(s_rows, 104, kc, lane);
    v16h b = load_b_glb(WmsgT + mode * OPAD * KP, KP, wave, kc, lane);
    acc = wmma32(a, b, acc);
  }
  const int o = wave * 16 + (lane & 15);
  if (o < STATE) {
    float bm = bmsg[mode * STATE + o];
#pragma unroll
    for (int r = 0; r < 8; ++r) {
      int m = r + ((lane >= 16) ? 8 : 0);
      unsigned enc = ordEncode(acc[r] + bm);
      atomicMax(&agg[s_dst[m] * OPAD + o], enc);
    }
  }
}

__global__ __launch_bounds__(256) void agg_to_cat_kernel(const unsigned* agg, f16* cat160) {
  int i = blockIdx.x * 256 + threadIdx.x;
  if (i >= NNODES * STATE) return;
  int node = i / STATE, k = i % STATE;
  cat160[node * CATK + STATE + k] = (f16)ordDecode(agg[node * OPAD + k]);
}

// ---------------- update: out = [x, agg] @ W_upd + b ----------------
__global__ __launch_bounds__(160) void upd_kernel(
    const f16* cat160, const f16* WupdT, const float* bupd, f16* out96)
{
  __shared__ __attribute__((aligned(16))) f16 s_rows[16 * 168];
  const int tid = threadIdx.x, lane = tid & 31, wave = tid >> 5;
  const int m0 = blockIdx.x * 16;

  for (int i = tid; i < 16 * CATK; i += 160) {
    int m = i / CATK, k = i % CATK;
    s_rows[m * 168 + k] = cat160[(m0 + m) * CATK + k];
  }
  __syncthreads();

  v8f acc = vzero8();
#pragma unroll
  for (int kc = 0; kc < 5; ++kc) {
    v16h a = load_a_lds(s_rows, 168, kc, lane);
    v16h b = load_b_glb(WupdT, CATK, wave, kc, lane);
    acc = wmma32(a, b, acc);
  }
  const int o = wave * 16 + (lane & 15);
  if (o < STATE) {
    float bb = bupd[o];
#pragma unroll
    for (int r = 0; r < 8; ++r) {
      int m = r + ((lane >= 16) ? 8 : 0);
      out96[(m0 + m) * KP + o] = (f16)(acc[r] + bb);
    }
  }
  for (int i = tid; i < 16 * (KP - STATE); i += 160) {
    int m = i / (KP - STATE), k = STATE + (i % (KP - STATE));
    out96[(m0 + m) * KP + k] = (f16)0.f;
  }
}

// ---------------- head: relu(out[t] @ W1 + b1) @ W2 + b2 ----------------
__global__ __launch_bounds__(160) void head_kernel(
    const f16* out96, const int* targets, const f16* Wlin1T,
    const float* blin1, const float* Wlin2, const float* blin2, float* d_out)
{
  __shared__ __attribute__((aligned(16))) f16 s_y[16 * 104];
  __shared__ __attribute__((aligned(16))) f16 s_y1[16 * 72];
  const int tid = threadIdx.x, lane = tid & 31, wave = tid >> 5;
  const int t0 = blockIdx.x * 16;

  for (int i = tid; i < 16 * KP; i += 160) {
    int m = i / KP, k = i % KP;
    s_y[m * 104 + k] = out96[targets[t0 + m] * KP + k];
  }
  __syncthreads();

  v8f acc = vzero8();
#pragma unroll
  for (int kc = 0; kc < 3; ++kc) {
    v16h a = load_a_lds(s_y, 104, kc, lane);
    v16h b = load_b_glb(Wlin1T, KP, wave, kc, lane);
    acc = wmma32(a, b, acc);
  }
  const int o = wave * 16 + (lane & 15);
  if (o < STATE) {
    float bb = blin1[o];
#pragma unroll
    for (int r = 0; r < 8; ++r) {
      int m = r + ((lane >= 16) ? 8 : 0);
      float v = acc[r] + bb;
      s_y1[m * 72 + o] = (f16)fmaxf(v, 0.f);
    }
  }
  __syncthreads();

  if (tid < 32) {
    int m = tid & 15, c = tid >> 4;
    float s = blin2[c];
    for (int k = 0; k < STATE; ++k)
      s += (float)s_y1[m * 72 + k] * Wlin2[k * 2 + c];
    d_out[(t0 + m) * 2 + c] = s;
  }
}

extern "C" void kernel_launch(void* const* d_in, const int* in_sizes, int n_in,
                              void* d_out, int out_size, void* d_ws, size_t ws_size,
                              hipStream_t stream) {
  const int*   tokens    = (const int*)d_in[0];
  const int*   node_type = (const int*)d_in[1];
  const int*   edges     = (const int*)d_in[2];
  const int*   targets   = (const int*)d_in[3];
  const float* embed     = (const float*)d_in[4];
  const float* Wih_map   = (const float*)d_in[5];
  const float* Whh_map   = (const float*)d_in[6];
  const float* bih_map   = (const float*)d_in[7];
  const float* bhh_map   = (const float*)d_in[8];
  const float* Wih_mem   = (const float*)d_in[9];
  const float* Whh_mem   = (const float*)d_in[10];
  const float* bih_mem   = (const float*)d_in[11];
  const float* bhh_mem   = (const float*)d_in[12];
  const float* Wmsg      = (const float*)d_in[13];
  const float* bmsg      = (const float*)d_in[14];
  const float* Wupd      = (const float*)d_in[15];
  const float* bupd      = (const float*)d_in[16];
  const float* Wlin1     = (const float*)d_in[17];
  const float* blin1     = (const float*)d_in[18];
  const float* Wlin2     = (const float*)d_in[19];
  const float* blin2     = (const float*)d_in[20];

  char* ws = (char*)d_ws;
  f16* gWih_map = (f16*)(ws + OFF_WIH_MAP);
  f16* gWhh_map = (f16*)(ws + OFF_WHH_MAP);
  f16* gWih_mem = (f16*)(ws + OFF_WIH_MEM);
  f16* gWhh_mem = (f16*)(ws + OFF_WHH_MEM);
  f16* gWmsgT   = (f16*)(ws + OFF_WMSGT);
  f16* gWupdT   = (f16*)(ws + OFF_WUPDT);
  f16* gWlin1T  = (f16*)(ws + OFF_WLIN1T);
  f16* x96      = (f16*)(ws + OFF_X96);
  f16* cat160   = (f16*)(ws + OFF_CAT);
  unsigned* agg = (unsigned*)(ws + OFF_AGG);
  f16* out96    = (f16*)(ws + OFF_OUT96);

  prep_kernel<<<(92672 + 255) / 256, 256, 0, stream>>>(
      Wih_map, Whh_map, Wih_mem, Whh_mem, Wmsg, Wupd, Wlin1,
      gWih_map, gWhh_map, gWih_mem, gWhh_mem, gWmsgT, gWupdT, gWlin1T);

  init_agg_kernel<<<(NNODES * OPAD + 255) / 256, 256, 0, stream>>>(agg);

  encode_kernel<<<NNODES / 16, 256, 0, stream>>>(
      tokens, node_type, embed,
      gWih_map, gWhh_map, gWih_mem, gWhh_mem,
      bih_map, bhh_map, bih_mem, bhh_mem, x96, cat160);

  msg_kernel<<<2 * (NEDGES / 16) + NNODES / 16, 160, 0, stream>>>(
      x96, gWmsgT, bmsg, edges, agg);

  agg_to_cat_kernel<<<(NNODES * STATE + 255) / 256, 256, 0, stream>>>(agg, cat160);

  upd_kernel<<<NNODES / 16, 160, 0, stream>>>(cat160, gWupdT, bupd, out96);

  head_kernel<<<NTGT / 16, 160, 0, stream>>>(
      out96, targets, gWlin1T, blin1, Wlin2, blin2, (float*)d_out);
}
